// VectorQuantizerEnt_38465727103637
// MI455X (gfx1250) — matrix operand order repaired and checked by reading, not compiled, not run on hardware
//
#include <hip/hip_runtime.h>
#include <hip/hip_bf16.h>
#include <math.h>

// ---------- problem constants ----------
#define M_TOTAL 16384      // B*T = 8*2048
#define DDIM    1024
#define KDIM    1024
#define GAMMA   1.0f
#define EPSV    1e-8f

#define MT      32                 // rows per workgroup in the GEMM kernel
#define NWG     (M_TOTAL / MT)     // 512 workgroups
#define LDSSTR  (KDIM + 8)         // padded LDS row stride (breaks bank conflicts)

// ---------- vector types (trivial ext-vectors so unions are legal) ----------
typedef __attribute__((ext_vector_type(16))) __bf16      v16bf;
typedef __attribute__((ext_vector_type(8)))  __bf16      v8bf;
typedef __attribute__((ext_vector_type(8)))  float       v8f;
typedef __attribute__((ext_vector_type(4)))  float       f32x4;
typedef __attribute__((ext_vector_type(4)))  unsigned int u32x4;

union BF16x16 { v16bf v; u32x4 u[2]; };

// ---------- workspace layout (float units) ----------
// [0]                : loss scalar
// [16 .. 16+512)     : per-WG entropy partials
// [1024 .. 1024+512*1024) : per-WG diversity partials
// after that         : bf16 codebook (KDIM*DDIM __bf16 = 2MB)
#define WS_ENT_OFF 16
#define WS_DIV_OFF 1024
#define WS_CB_OFF  (WS_DIV_OFF + NWG * KDIM)

// =====================================================================
// k0: convert f32 codebook -> bf16 (read once; halves GEMM B-side bytes)
// =====================================================================
__global__ __launch_bounds__(256) void cvt_cb_kernel(const float* __restrict__ cb,
                                                     __bf16* __restrict__ out) {
    size_t i = ((size_t)blockIdx.x * 256 + threadIdx.x) * 8;
    f32x4 a = *(const f32x4*)(cb + i);
    f32x4 b = *(const f32x4*)(cb + i + 4);
    v8bf  o;
#pragma unroll
    for (int e = 0; e < 4; ++e) o[e]     = (__bf16)a[e];
#pragma unroll
    for (int e = 0; e < 4; ++e) o[e + 4] = (__bf16)b[e];
    *(v8bf*)(out + i) = o;
}

// =====================================================================
// k1: fused bf16-WMMA GEMM (similarity) + softmax stats per 32-row tile
// =====================================================================
__device__ inline v16bf load_a_frag(const float* p) {
    // A 16x32 bf16 layout: lanes 0-15 hold d = {0..7, 16..23}, lanes 16-31 +8
    // (p already includes the per-halfwave +8 element offset)
    f32x4 x0 = *(const f32x4*)(p);
    f32x4 x1 = *(const f32x4*)(p + 4);
    f32x4 x2 = *(const f32x4*)(p + 16);
    f32x4 x3 = *(const f32x4*)(p + 20);
    v16bf r;
#pragma unroll
    for (int e = 0; e < 4; ++e) {
        r[e]      = (__bf16)x0[e];
        r[e + 4]  = (__bf16)x1[e];
        r[e + 8]  = (__bf16)x2[e];
        r[e + 12] = (__bf16)x3[e];
    }
    return r;
}

__global__ __launch_bounds__(256) void gemm_softmax_kernel(
        const float* __restrict__ inp,      // [M_TOTAL, DDIM] f32
        const __bf16* __restrict__ cbb,     // [KDIM, DDIM] bf16
        float* __restrict__ ws_ent,         // [NWG]
        float* __restrict__ ws_div,         // [NWG, KDIM]
        int* __restrict__ idx_out)          // [M_TOTAL]
{
    __shared__ float s_sim[MT][LDSSTR];     // ~129 KB of 320 KB WGP LDS
    __shared__ float s_red[MT];

    const int tid  = threadIdx.x;
    const int wave = tid >> 5;
    const int lane = tid & 31;
    const int l15  = lane & 15;
    const int hi   = lane >> 4;             // half-wave select
    const int m_base = blockIdx.x * MT;

    // ---- accumulators: 2 M-subtiles x 8 N-tiles of 16x16 f32 ----
    v8f acc[2][8];
#pragma unroll
    for (int s = 0; s < 2; ++s)
#pragma unroll
        for (int t = 0; t < 8; ++t)
#pragma unroll
            for (int r = 0; r < 8; ++r) acc[s][t][r] = 0.0f;

    const int dselA = hi ? 8 : 0;                                // A interleave
    const float* pa0 = inp + (size_t)(m_base + l15) * DDIM + dselA;
    const float* pa1 = pa0 + (size_t)16 * DDIM;
    const int dselB = hi ? 8 : 0;                                // B: same interleave as A
    const __bf16* pbb = cbb + (size_t)(wave * 128 + l15) * DDIM + dselB;

    for (int d0 = 0; d0 < DDIM; d0 += 32) {
        v16bf a0 = load_a_frag(pa0 + d0);
        v16bf a1 = load_a_frag(pa1 + d0);
#pragma unroll
        for (int t = 0; t < 8; ++t) {
            const __bf16* pb = pbb + (size_t)t * 16 * DDIM + d0;
            BF16x16 bf;
            bf.u[0] = *(const u32x4*)(pb);
            bf.u[1] = *(const u32x4*)(pb + 16);
            acc[0][t] = __builtin_amdgcn_wmma_f32_16x16x32_bf16(
                false, a0, false, bf.v, (short)0, acc[0][t], false, false);
            acc[1][t] = __builtin_amdgcn_wmma_f32_16x16x32_bf16(
                false, a1, false, bf.v, (short)0, acc[1][t], false, false);
        }
    }

    // ---- spill similarity tile to LDS (row = M, col = K) ----
#pragma unroll
    for (int s = 0; s < 2; ++s)
#pragma unroll
        for (int t = 0; t < 8; ++t)
#pragma unroll
            for (int r = 0; r < 8; ++r)
                s_sim[s * 16 + hi * 8 + r][wave * 128 + t * 16 + l15] = acc[s][t][r];
    __syncthreads();

    // ---- per-row softmax stats: 8 threads per row ----
    const int rowg = tid >> 3;              // 0..31
    const int j    = tid & 7;

    // max + argmax (first-max tiebreak)
    float mx = -3.0e38f; int mi = 0;
    for (int c = j; c < KDIM; c += 8) {
        float v = s_sim[rowg][c];
        if (v > mx) { mx = v; mi = c; }
    }
#pragma unroll
    for (int off = 1; off < 8; off <<= 1) {
        float om = __shfl_xor(mx, off, 32);
        int   oi = __shfl_xor(mi, off, 32);
        if (om > mx || (om == mx && oi < mi)) { mx = om; mi = oi; }
    }

    // sum of exp (store exp back into LDS)
    float se = 0.0f;
    for (int c = j; c < KDIM; c += 8) {
        float e = __expf(s_sim[rowg][c] - mx);
        s_sim[rowg][c] = e;
        se += e;
    }
#pragma unroll
    for (int off = 1; off < 8; off <<= 1) se += __shfl_xor(se, off, 32);
    float invZ = 1.0f / se;

    // probabilities + entropy term sum(p*log2(p+eps))
    float ent = 0.0f;
    for (int c = j; c < KDIM; c += 8) {
        float p = s_sim[rowg][c] * invZ;
        s_sim[rowg][c] = p;
        ent += p * __log2f(p + EPSV);
    }
#pragma unroll
    for (int off = 1; off < 8; off <<= 1) ent += __shfl_xor(ent, off, 32);

    if (j == 0) {
        idx_out[m_base + rowg] = mi;
        s_red[rowg] = ent;
    }
    __syncthreads();

    if (tid == 0) {
        float s = 0.0f;
#pragma unroll
        for (int r = 0; r < MT; ++r) s += s_red[r];
        ws_ent[blockIdx.x] = s;             // per-WG partial (deterministic)
    }

    // ---- diversity partials: sum of p over the 32 local rows, per column ----
#pragma unroll
    for (int i = 0; i < 4; ++i) {
        int c = tid + i * 256;
        float s = 0.0f;
#pragma unroll
        for (int r = 0; r < MT; ++r) s += s_sim[r][c];
        ws_div[(size_t)blockIdx.x * KDIM + c] = s;
    }
}

// =====================================================================
// k2: single-block reduction -> loss scalar
// =====================================================================
__global__ __launch_bounds__(1024) void finalize_kernel(
        const float* __restrict__ ws_ent,
        const float* __restrict__ ws_div,
        float* __restrict__ loss_out)
{
    __shared__ float s_part[32];
    const int k = threadIdx.x;              // one column of K per thread
    float dv = 0.0f;
    for (int w = 0; w < NWG; ++w) dv += ws_div[(size_t)w * KDIM + k];
    dv *= (1.0f / (float)M_TOTAL);
    float dterm = dv * __log2f(dv + EPSV);
#pragma unroll
    for (int off = 16; off > 0; off >>= 1) dterm += __shfl_xor(dterm, off, 32);
    if ((k & 31) == 0) s_part[k >> 5] = dterm;
    __syncthreads();
    if (k == 0) {
        float dsum = 0.0f;
#pragma unroll
        for (int w = 0; w < 32; ++w) dsum += s_part[w];
        float es = 0.0f;
        for (int w = 0; w < NWG; ++w) es += ws_ent[w];
        float h_clust = -es / (float)M_TOTAL;
        float h_div   = -dsum;
        loss_out[0] = h_clust - GAMMA * h_div;
    }
}

// =====================================================================
// k3: fill quantization_loss [B,T,K] with the scalar
// =====================================================================
__global__ __launch_bounds__(256) void fill_loss_kernel(float* __restrict__ out,
                                                        const float* __restrict__ loss) {
    float v = loss[0];
    size_t i = ((size_t)blockIdx.x * 256 + threadIdx.x) * 4;
    f32x4 f = {v, v, v, v};
    *(f32x4*)(out + i) = f;
}

// =====================================================================
// k4: gather nearest codebook row, mean-center, L2-normalize
// =====================================================================
__global__ __launch_bounds__(256) void quantize_kernel(
        const float* __restrict__ cb,
        const int* __restrict__ idx,
        float* __restrict__ out)
{
    __shared__ float s_part[8];
    const int row = blockIdx.x;
    const int k   = idx[row];
    const float* src = cb + (size_t)k * DDIM;
    f32x4 v = *(const f32x4*)(src + threadIdx.x * 4);

    float t = v[0] + v[1] + v[2] + v[3];
#pragma unroll
    for (int off = 16; off > 0; off >>= 1) t += __shfl_xor(t, off, 32);
    if ((threadIdx.x & 31) == 0) s_part[threadIdx.x >> 5] = t;
    __syncthreads();
    float total = 0.0f;
#pragma unroll
    for (int i = 0; i < 8; ++i) total += s_part[i];
    float mean = total * (1.0f / (float)DDIM);

    f32x4 c;
#pragma unroll
    for (int e = 0; e < 4; ++e) c[e] = v[e] - mean;
    float sq = c[0]*c[0] + c[1]*c[1] + c[2]*c[2] + c[3]*c[3];
    __syncthreads();                        // s_part reuse
#pragma unroll
    for (int off = 16; off > 0; off >>= 1) sq += __shfl_xor(sq, off, 32);
    if ((threadIdx.x & 31) == 0) s_part[threadIdx.x >> 5] = sq;
    __syncthreads();
    float ssq = 0.0f;
#pragma unroll
    for (int i = 0; i < 8; ++i) ssq += s_part[i];
    float rn = rsqrtf(ssq);

    f32x4 o;
#pragma unroll
    for (int e = 0; e < 4; ++e) o[e] = c[e] * rn;
    *(f32x4*)(out + (size_t)row * DDIM + threadIdx.x * 4) = o;
}

// =====================================================================
// k5: codebook passthrough
// =====================================================================
__global__ __launch_bounds__(256) void copy_cb_kernel(const float* __restrict__ cb,
                                                      float* __restrict__ out) {
    size_t i = ((size_t)blockIdx.x * 256 + threadIdx.x) * 4;
    *(f32x4*)(out + i) = *(const f32x4*)(cb + i);
}

// =====================================================================
extern "C" void kernel_launch(void* const* d_in, const int* in_sizes, int n_in,
                              void* d_out, int out_size, void* d_ws, size_t ws_size,
                              hipStream_t stream) {
    const float* inp = (const float*)d_in[0];   // [8,2048,1024] f32
    const float* cb  = (const float*)d_in[1];   // [1024,1024]  f32

    float* out      = (float*)d_out;
    float* q_out    = out;                              // quantized  [B,T,D]
    float* loss_out = out + (size_t)M_TOTAL * DDIM;     // loss       [B,T,K]
    int*   idx_out  = (int*)(out + (size_t)M_TOTAL * DDIM + (size_t)M_TOTAL * KDIM);
    float* cb_out   = out + (size_t)M_TOTAL * DDIM + (size_t)M_TOTAL * KDIM + M_TOTAL;

    float*  wsf     = (float*)d_ws;
    float*  ws_loss = wsf;
    float*  ws_ent  = wsf + WS_ENT_OFF;
    float*  ws_div  = wsf + WS_DIV_OFF;
    __bf16* ws_cb   = (__bf16*)(wsf + WS_CB_OFF);

    // k0: codebook f32 -> bf16 (1M elems / 8 per thread)
    cvt_cb_kernel<<<KDIM * DDIM / (256 * 8), 256, 0, stream>>>(cb, ws_cb);

    // k1: fused WMMA GEMM + softmax statistics
    gemm_softmax_kernel<<<NWG, 256, 0, stream>>>(inp, ws_cb, ws_ent, ws_div, idx_out);

    // k2: loss scalar
    finalize_kernel<<<1, 1024, 0, stream>>>(ws_ent, ws_div, ws_loss);

    // k3: broadcast-fill quantization_loss (16M floats, 4/thread)
    fill_loss_kernel<<<(size_t)M_TOTAL * KDIM / (256 * 4), 256, 0, stream>>>(loss_out, ws_loss);

    // k4: gather + center + normalize quantized
    quantize_kernel<<<M_TOTAL, 256, 0, stream>>>(cb, idx_out, q_out);

    // k5: codebook passthrough
    copy_cb_kernel<<<KDIM * DDIM / (256 * 4), 256, 0, stream>>>(cb, cb_out);
}
